// LocalSpatialAttentionPatchKeras_58119497450444
// MI455X (gfx1250) — matrix operand order repaired and checked by reading, not compiled, not run on hardware
//
#include <hip/hip_runtime.h>
#include <hip/hip_bf16.h>

typedef __attribute__((ext_vector_type(16))) __bf16 v16bf;
typedef __attribute__((ext_vector_type(8)))  __bf16 v8bf;
typedef __attribute__((ext_vector_type(8)))  float  v8f;

#define BATCH 4
#define HH 64
#define WW 64
#define CC 256
#define SS 25
#define PIX (BATCH*HH*WW)   // 16384 pixels
#define KK 256              // GEMM inner dim (C)

// ---------------- Stage 0: f32 -> bf16 conversion of x ----------------
__global__ void cvt_x_kernel(const float* __restrict__ x, __bf16* __restrict__ xb, int n) {
  int i = (blockIdx.x * blockDim.x + threadIdx.x) * 4;
  if (i < n) {
    float4 v = *(const float4*)(x + i);
    xb[i + 0] = (__bf16)v.x;
    xb[i + 1] = (__bf16)v.y;
    xb[i + 2] = (__bf16)v.z;
    xb[i + 3] = (__bf16)v.w;
  }
}

// ---------------- Stage 0b: pack W^T (bf16) + fused qkv bias ----------------
// wt[o][c] = W_{q|k|v}[c][o&255]   (o in 0..767: 0-255 q, 256-511 k, 512-767 v)
__global__ void pack_wqkv_kernel(const float* __restrict__ Wq, const float* __restrict__ Wk,
                                 const float* __restrict__ Wv,
                                 const float* __restrict__ bq, const float* __restrict__ bk,
                                 const float* __restrict__ bv,
                                 __bf16* __restrict__ wt, float* __restrict__ bias768) {
  int idx = blockIdx.x * blockDim.x + threadIdx.x;   // 768*256 threads
  int o = idx >> 8, c = idx & 255;
  const float* W = (o < 256) ? Wq : ((o < 512) ? Wk : Wv);
  wt[(size_t)o * 256 + c] = (__bf16)W[(size_t)c * 256 + (o & 255)];
  if (idx < 768) {
    bias768[idx] = (idx < 256) ? bq[idx] : ((idx < 512) ? bk[idx - 256] : bv[idx - 512]);
  }
}

// wo_t[cout][nd] = Wo[nd][cout]  (Wo flat is (NH*KD, C) row-major)
__global__ void pack_wo_kernel(const float* __restrict__ Wo, __bf16* __restrict__ wot) {
  int idx = blockIdx.x * blockDim.x + threadIdx.x;   // 256*256 threads
  int cout = idx >> 8, nd = idx & 255;
  wot[(size_t)cout * 256 + nd] = (__bf16)Wo[(size_t)nd * 256 + cout];
}

// ---------------- Stage 0c: pos_emb projections (tiny, f32 exact) ----------------
__global__ void pos_proj_kernel(const float* __restrict__ pe,
                                const float* __restrict__ Wq, const float* __restrict__ Wk,
                                float* __restrict__ pq, float* __restrict__ pk) {
  int s = blockIdx.x;          // 25 blocks
  int o = threadIdx.x;         // 256 threads
  const float* per = pe + s * 256;
  float aq = 0.f, ak = 0.f;
  for (int c = 0; c < 256; ++c) {
    float v = per[c];
    aq = fmaf(v, Wq[(size_t)c * 256 + o], aq);
    ak = fmaf(v, Wk[(size_t)c * 256 + o], ak);
  }
  pq[s * 256 + o] = aq;
  pk[s * 256 + o] = ak;
}

// ---------------- WMMA bf16 GEMM: C[M x N] = A[M x K] * Bt[N x K]^T + bias ----------------
// Register-blocked: each wave computes a 32x64 block (2 M-tiles x 4 N-tiles, 8 v8f accs).
// Block = 4 waves -> covers M=32, N=256. Per K-step: 12 b128 loads feed 8 WMMAs.
// A-fragment (ISA 7.12.2): lane L -> row (L&15), 16B chunks at k=(L>>4)*8 and 16+(L>>4)*8
// B-fragment: lane L -> col (L&15), 16B chunks at k=(L>>4)*8 and 16+(L>>4)*8 (from transposed Bt)
__global__ void wmma_gemm_bf16_kernel(const __bf16* __restrict__ A,
                                      const __bf16* __restrict__ Bt,
                                      const float* __restrict__ bias,
                                      float* __restrict__ C, int N) {
  const int lane = threadIdx.x & 31;
  const int wave = threadIdx.x >> 5;                 // 0..3
  const int tileM = blockIdx.x * 32;
  const int tileN = blockIdx.y * 256 + wave * 64;
  const int half = lane >> 4;                        // 0 or 1

  const __bf16* arow0 = A + (size_t)(tileM + (lane & 15)) * KK + half * 8;
  const __bf16* arow1 = arow0 + (size_t)16 * KK;
  const __bf16* brow[4];
#pragma unroll
  for (int j = 0; j < 4; ++j)
    brow[j] = Bt + (size_t)(tileN + j * 16 + (lane & 15)) * KK + half * 8;

  v8f acc[2][4] = {};
#pragma unroll
  for (int k = 0; k < KK; k += 32) {
    v8bf a0lo = *(const v8bf*)(arow0 + k);
    v8bf a0hi = *(const v8bf*)(arow0 + k + 16);
    v8bf a1lo = *(const v8bf*)(arow1 + k);
    v8bf a1hi = *(const v8bf*)(arow1 + k + 16);
    v16bf a0 = __builtin_shufflevector(a0lo, a0hi, 0,1,2,3,4,5,6,7,8,9,10,11,12,13,14,15);
    v16bf a1 = __builtin_shufflevector(a1lo, a1hi, 0,1,2,3,4,5,6,7,8,9,10,11,12,13,14,15);
#pragma unroll
    for (int j = 0; j < 4; ++j) {
      v8bf b0 = *(const v8bf*)(brow[j] + k);
      v8bf b1 = *(const v8bf*)(brow[j] + k + 16);
      v16bf b = __builtin_shufflevector(b0, b1, 0,1,2,3,4,5,6,7,8,9,10,11,12,13,14,15);
      acc[0][j] = __builtin_amdgcn_wmma_f32_16x16x32_bf16(false, a0, false, b, (short)0,
                                                          acc[0][j], false, false);
      acc[1][j] = __builtin_amdgcn_wmma_f32_16x16x32_bf16(false, a1, false, b, (short)0,
                                                          acc[1][j], false, false);
    }
  }

#pragma unroll
  for (int j = 0; j < 4; ++j) {
    const int cn = tileN + j * 16 + (lane & 15);
    const float bvj = bias[cn];
#pragma unroll
    for (int mi = 0; mi < 2; ++mi) {
      const int rbase = tileM + mi * 16 + half * 8;
#pragma unroll
      for (int r = 0; r < 8; ++r) {
        C[(size_t)(rbase + r) * N + cn] = acc[mi][j][r] + bvj;
      }
    }
  }
}

// ---------------- Attention stage (f32 VALU, one wave per pixel) ----------------
// xqkv: [PIX][768] f32 (q+bq | k+bk | v+bv). Lane L owns channels [8L, 8L+8); head = L/4.
__global__ void attn_kernel(const float* __restrict__ xqkv,
                            const float* __restrict__ pq,
                            const float* __restrict__ pk,
                            __bf16* __restrict__ ohb) {
  const int lane = threadIdx.x & 31;
  const int wave = threadIdx.x >> 5;
  const int p = blockIdx.x * 8 + wave;               // pixel index
  const int b = p >> 12;                             // / (64*64)
  const int yx = p & 4095;
  const int y = yx >> 6, x = yx & 63;

  // shifted (not clamped) 5x5 window per reference's _patch_row_index
  const int ys = min(max(y, 2), HH - 3) - 2;
  const int xs = min(max(x, 2), WW - 3) - 2;
  const int qidx = (y - ys) * 5 + (x - xs);

  const int c0 = lane * 8;
  const float* qrow  = xqkv + (size_t)p * 768;
  const float* pqrow = pq + qidx * 256 + c0;

  float q[8];
#pragma unroll
  for (int i = 0; i < 8; ++i) q[i] = qrow[c0 + i] + pqrow[i];

  const float scale = 0.17677669529663687f;          // 1/sqrt(32)
  float sc[SS];
#pragma unroll
  for (int s = 0; s < SS; ++s) {
    const int ny = ys + s / 5, nx = xs + s % 5;
    const size_t np = ((size_t)b << 12) | (ny << 6) | nx;
    const float* krow  = xqkv + np * 768 + 256 + c0;
    const float* pkrow = pk + s * 256 + c0;
    float part = 0.f;
#pragma unroll
    for (int i = 0; i < 8; ++i) part = fmaf(q[i], krow[i] + pkrow[i], part);
    // reduce 4 lanes of the same head (lanes 4n..4n+3)
    part += __shfl_xor(part, 1, 32);
    part += __shfl_xor(part, 2, 32);
    sc[s] = part * scale;
  }

  float mx = sc[0];
#pragma unroll
  for (int s = 1; s < SS; ++s) mx = fmaxf(mx, sc[s]);
  float sum = 0.f;
#pragma unroll
  for (int s = 0; s < SS; ++s) { sc[s] = __expf(sc[s] - mx); sum += sc[s]; }
  const float inv = 1.f / sum;

  float o[8] = {0.f, 0.f, 0.f, 0.f, 0.f, 0.f, 0.f, 0.f};
#pragma unroll
  for (int s = 0; s < SS; ++s) {
    const int ny = ys + s / 5, nx = xs + s % 5;
    const size_t np = ((size_t)b << 12) | (ny << 6) | nx;
    const float* vrow = xqkv + np * 768 + 512 + c0;
    const float a = sc[s] * inv;
#pragma unroll
    for (int i = 0; i < 8; ++i) o[i] = fmaf(a, vrow[i], o[i]);
  }

  __bf16* orow = ohb + (size_t)p * 256 + c0;
#pragma unroll
  for (int i = 0; i < 8; ++i) orow[i] = (__bf16)o[i];
}

// ---------------- Launch ----------------
extern "C" void kernel_launch(void* const* d_in, const int* in_sizes, int n_in,
                              void* d_out, int out_size, void* d_ws, size_t ws_size,
                              hipStream_t stream) {
  const float* x  = (const float*)d_in[0];
  const float* pe = (const float*)d_in[1];
  const float* Wq = (const float*)d_in[2];
  const float* bq = (const float*)d_in[3];
  const float* Wk = (const float*)d_in[4];
  const float* bk = (const float*)d_in[5];
  const float* Wv = (const float*)d_in[6];
  const float* bv = (const float*)d_in[7];
  const float* Wo = (const float*)d_in[8];
  const float* bo = (const float*)d_in[9];
  float* out = (float*)d_out;
  (void)in_sizes; (void)n_in; (void)out_size; (void)ws_size;

  char* ws = (char*)d_ws;
  size_t off = 0;
  auto take = [&](size_t bytes) -> char* {
    char* ptr = ws + off;
    off = (off + bytes + 255) & ~(size_t)255;
    return ptr;
  };
  __bf16* xb      = (__bf16*)take((size_t)PIX * CC * 2);       // 8 MB
  __bf16* wt_qkv  = (__bf16*)take((size_t)768 * 256 * 2);      // 384 KB
  __bf16* wo_t    = (__bf16*)take((size_t)256 * 256 * 2);      // 128 KB
  float*  bias768 = (float*) take((size_t)768 * 4);
  float*  pqb     = (float*) take((size_t)SS * CC * 4);
  float*  pkb     = (float*) take((size_t)SS * CC * 4);
  float*  xqkv    = (float*) take((size_t)PIX * 768 * 4);      // 48 MB
  __bf16* ohb     = (__bf16*)take((size_t)PIX * CC * 2);       // 8 MB

  // Stage 0: conversions / packing (tiny)
  cvt_x_kernel<<<(PIX * CC / 4 + 255) / 256, 256, 0, stream>>>(x, xb, PIX * CC);
  pack_wqkv_kernel<<<768, 256, 0, stream>>>(Wq, Wk, Wv, bq, bk, bv, wt_qkv, bias768);
  pack_wo_kernel<<<256, 256, 0, stream>>>(Wo, wo_t);
  pos_proj_kernel<<<SS, 256, 0, stream>>>(pe, Wq, Wk, pqb, pkb);

  // Stage 1: fused QKV projection GEMM  [16384 x 256] * [256 x 768]
  {
    dim3 grid(PIX / 32, 768 / 256);
    wmma_gemm_bf16_kernel<<<grid, 128, 0, stream>>>(xb, wt_qkv, bias768, xqkv, 768);
  }

  // Stage 2: local attention, f32 VALU, wave-per-pixel
  attn_kernel<<<PIX / 8, 256, 0, stream>>>(xqkv, pqb, pkb, ohb);

  // Stage 3: output projection GEMM  [16384 x 256] * [256 x 256] (+bo) -> d_out
  {
    dim3 grid(PIX / 32, 256 / 256);
    wmma_gemm_bf16_kernel<<<grid, 128, 0, stream>>>(ohb, wo_t, bo, out, 256);
  }
}
